// HTR_Capsule_35656818491747
// MI455X (gfx1250) — compile-verified, matrix operand order checked by reading, not run on hardware
//
#include <hip/hip_runtime.h>
#include <hip/hip_bf16.h>

// ---------------------------------------------------------------------------
// HTR capsule forward for MI455X (gfx1250, wave32).
// Vote GEMMs: v_wmma_f32_16x16x32_f16 with LDS-staged, fragment-swizzled B
// tiles (loaded once per block) and float4-loaded A fragments reused across
// all output capsules. Routing/MHA are streaming VALU/LDS kernels.
// Total HBM traffic ~0.35 GB -> ~15 us at 23.3 TB/s; compute is trivial.
// ---------------------------------------------------------------------------

typedef __attribute__((ext_vector_type(16))) _Float16 v16h;
typedef __attribute__((ext_vector_type(8)))  float    v8f;

#define EPS_SQ   1e-7f
#define LN_EPS   1e-6f

// WMMA f16 A/B fragment mapping (16x32, K-major pairs):
//   lane l holds M/N = l%16, half = l/16; element e (0..15) -> K = (e<8 ? e : e+8) + 8*half
// Inverse for staging (K < 16): lane = mn + 16*(k>>3), e = k&7; e>=8 slots stay zero.

// ---------------------------------------------------------------------------
// votes0[b,i,o,j] = sum_{k<8} W0[i,o,j,k] * x[b,i,k]
// One block per input capsule i; W0[i] staged to LDS in fragment layout once;
// 8 waves tile the 128-batch; each wave runs 5 WMMAs (one per o) on one A frag.
// ---------------------------------------------------------------------------
__global__ __launch_bounds__(256) void k_votes0(const float* __restrict__ x,
                                                const float* __restrict__ W0,
                                                float* __restrict__ votes0) {
  const int i = blockIdx.x;                       // 0..1151
  __shared__ __align__(32) _Float16 bfrag[5 * 32 * 16];   // [o][lane][e], 5 KB

  for (int t = threadIdx.x; t < 5 * 32 * 16; t += 256) bfrag[t] = (_Float16)0.f;
  __syncthreads();
  // Stage W0[i] (k fastest -> coalesced): k<8 => lane = j (half 0), e = k.
  for (int t = threadIdx.x; t < 5 * 16 * 8; t += 256) {
    const int k = t & 7, j = (t >> 3) & 15, o = t >> 7;
    bfrag[(o * 32 + j) * 16 + k] = (_Float16)W0[(size_t)i * 640 + t];
  }
  __syncthreads();

  const int wave = threadIdx.x >> 5;              // batch tile 0..7
  const int lane = threadIdx.x & 31;
  const int half = lane >> 4;
  const int mn   = lane & 15;
  const int brow = wave * 16 + mn;

  // A fragment: row = batch brow, real K depth 8 -> only half-0 lanes non-zero.
  v16h a{};
  if (half == 0) {
    const float4* xr = reinterpret_cast<const float4*>(&x[((size_t)brow * 1152 + i) * 8]);
    const float4 p0 = xr[0], p1 = xr[1];
    a[0] = (_Float16)p0.x; a[1] = (_Float16)p0.y; a[2] = (_Float16)p0.z; a[3] = (_Float16)p0.w;
    a[4] = (_Float16)p1.x; a[5] = (_Float16)p1.y; a[6] = (_Float16)p1.z; a[7] = (_Float16)p1.w;
  }

  const size_t obase = (size_t)(wave * 16) * 92160 + (size_t)i * 80;  // 92160 = 1152*5*16
#pragma unroll
  for (int o = 0; o < 5; ++o) {
    const v16h bf = *reinterpret_cast<const v16h*>(&bfrag[(o * 32 + lane) * 16]);
    v8f c{};
    c = __builtin_amdgcn_wmma_f32_16x16x32_f16(false, a, false, bf,
                                               (short)0, c, false, false);
    float* dst = &votes0[obase + (size_t)o * 16 + mn + (size_t)(8 * half) * 92160];
#pragma unroll
    for (int v = 0; v < 8; ++v)                   // constant row offsets -> ioffset
      dst[(size_t)v * 92160] = c[v];
  }
}

// ---------------------------------------------------------------------------
// votes1[b,i,o,j] = sum_{k<16} W1[i,o,j,k] * in1[b,i,k]
// in1 = concat(x.reshape(B,576,16), out0) gathered on the fly.
// One block per i; W1[i] (25.6 KB) staged to LDS fragment layout once;
// each wave: one float4-pair A load, 25 WMMAs.
// ---------------------------------------------------------------------------
__global__ __launch_bounds__(256) void k_votes1(const float* __restrict__ x,
                                                const float* __restrict__ out0,
                                                const float* __restrict__ W1,
                                                float* __restrict__ votes1) {
  const int i = blockIdx.x;                       // 0..580
  __shared__ __align__(32) _Float16 bfrag[25 * 32 * 16];  // 25.6 KB

  for (int t = threadIdx.x; t < 25 * 32 * 16; t += 256) bfrag[t] = (_Float16)0.f;
  __syncthreads();
  for (int t = threadIdx.x; t < 25 * 16 * 16; t += 256) {
    const int k = t & 15, j = (t >> 4) & 15, o = t >> 8;
    const int fl = j + 16 * (k >> 3);             // fragment lane
    bfrag[(o * 32 + fl) * 16 + (k & 7)] = (_Float16)W1[(size_t)i * 6400 + t];
  }
  __syncthreads();

  const int wave = threadIdx.x >> 5;
  const int lane = threadIdx.x & 31;
  const int half = lane >> 4;
  const int mn   = lane & 15;
  const int brow = wave * 16 + mn;

  // A fragment: K = 8*half .. 8*half+7 in elements 0..7; K 16..31 padded zero.
  v16h a{};
  {
    const float* ar = (i < 576)
        ? &x[(size_t)brow * 9216 + (size_t)i * 16 + 8 * half]
        : &out0[((size_t)brow * 5 + (i - 576)) * 16 + 8 * half];
    const float4* a4 = reinterpret_cast<const float4*>(ar);
    const float4 p0 = a4[0], p1 = a4[1];
    a[0] = (_Float16)p0.x; a[1] = (_Float16)p0.y; a[2] = (_Float16)p0.z; a[3] = (_Float16)p0.w;
    a[4] = (_Float16)p1.x; a[5] = (_Float16)p1.y; a[6] = (_Float16)p1.z; a[7] = (_Float16)p1.w;
  }

  const size_t obase = (size_t)(wave * 16) * 232400 + (size_t)i * 400;  // 232400 = 581*25*16
#pragma unroll 5
  for (int o = 0; o < 25; ++o) {
    const v16h bf = *reinterpret_cast<const v16h*>(&bfrag[(o * 32 + lane) * 16]);
    v8f c{};
    c = __builtin_amdgcn_wmma_f32_16x16x32_f16(false, a, false, bf,
                                               (short)0, c, false, false);
    float* dst = &votes1[obase + (size_t)o * 16 + mn + (size_t)(8 * half) * 232400];
#pragma unroll
    for (int v = 0; v < 8; ++v)
      dst[(size_t)v * 232400] = c[v];
  }
}

// ---------------------------------------------------------------------------
// Routing iteration 0: raw=0 -> uniform weights. One wave per (b,o):
// lanes 0-15 own d, half-waves split the i-range, squash via lane shuffles.
// ---------------------------------------------------------------------------
template <int NIN, int NOUT>
__global__ __launch_bounds__(256) void k_route_s0(const float* __restrict__ votes,
                                                  float* __restrict__ vout) {
  const int gw   = (int)((blockIdx.x * blockDim.x + threadIdx.x) >> 5);
  const int lane = threadIdx.x & 31;
  const int b = gw / NOUT, o = gw % NOUT;
  if (b >= 128) return;
  const int half = lane >> 4, d = lane & 15;

  float s = 0.f;
  for (int i = half; i < NIN; i += 2)
    s += votes[(((size_t)b * NIN + i) * NOUT + o) * 16 + d];
  s += __shfl_xor(s, 16);
  s *= (1.0f / (float)NOUT);

  float sq = s * s;                       // sum over d within 16-lane group
  sq += __shfl_xor(sq, 1);
  sq += __shfl_xor(sq, 2);
  sq += __shfl_xor(sq, 4);
  sq += __shfl_xor(sq, 8);
  const float v = (sq / (1.f + sq)) * s * rsqrtf(sq + EPS_SQ);
  if (half == 0) vout[((size_t)b * NOUT + o) * 16 + d] = v;
}

// raw[b,i,o] = dot(votes[b,i,o,:], vbar[b,o,:])  (initial raw is zero)
template <int NIN, int NOUT>
__global__ __launch_bounds__(256) void k_raw(const float* __restrict__ votes,
                                             const float* __restrict__ vbar,
                                             float* __restrict__ raw) {
  const size_t idx = (size_t)blockIdx.x * blockDim.x + threadIdx.x;
  if (idx >= (size_t)128 * NIN * NOUT) return;
  const int o = (int)(idx % NOUT);
  const size_t t = idx / NOUT;
  const int b = (int)(t / NIN);
  const float* vr = &votes[idx * 16];
  const float* vb = &vbar[((size_t)b * NOUT + o) * 16];
  float acc = 0.f;
#pragma unroll
  for (int d = 0; d < 16; ++d) acc += vr[d] * vb[d];
  raw[idx] = acc;
}

// ---------------------------------------------------------------------------
// Final routing pass: block per batch. Threads = 16 i-groups x 16 d-lanes.
// Per-thread register accumulators, LDS-atomic reduce, fused squash.
// MASKED applies the taxonomy soft-mask (level 1 only).
// ---------------------------------------------------------------------------
template <int NIN, int NOUT, bool MASKED>
__global__ __launch_bounds__(256) void k_route_final(const float* __restrict__ votes,
                                                     const float* __restrict__ raw,
                                                     const float* __restrict__ soft,
                                                     float* __restrict__ vout) {
  const int b = blockIdx.x;
  const int d = threadIdx.x & 15;
  const int g = threadIdx.x >> 4;         // 0..15
  __shared__ float s[NOUT * 16];
  for (int t = threadIdx.x; t < NOUT * 16; t += blockDim.x) s[t] = 0.f;
  __syncthreads();

  float acc[NOUT];
#pragma unroll
  for (int o = 0; o < NOUT; ++o) acc[o] = 0.f;

  for (int i = g; i < NIN; i += 16) {
    float logit[NOUT];
    const float* rp = &raw[((size_t)b * NIN + i) * NOUT];
    float mx = -1e30f;
#pragma unroll
    for (int o = 0; o < NOUT; ++o) {
      float l = rp[o];
      if (MASKED) {
        const int p = (i < 580) ? (i % 5) : 0;             // tile + remainder
        l *= soft[((size_t)b * 5 + p) * NOUT + o] * 0.5f;  // MASK_TEMP
      }
      logit[o] = l;
      mx = fmaxf(mx, l);
    }
    float sum = 0.f;
#pragma unroll
    for (int o = 0; o < NOUT; ++o) { logit[o] = __expf(logit[o] - mx); sum += logit[o]; }
    const float inv = 1.f / sum;
    const float* vp = &votes[(((size_t)b * NIN + i) * NOUT) * 16 + d];
#pragma unroll
    for (int o = 0; o < NOUT; ++o) acc[o] += logit[o] * inv * vp[o * 16];
  }
#pragma unroll
  for (int o = 0; o < NOUT; ++o) atomicAdd(&s[o * 16 + d], acc[o]);
  __syncthreads();

  if (threadIdx.x < NOUT) {               // squash per output capsule
    const int o = threadIdx.x;
    float sq = 0.f;
#pragma unroll
    for (int dd = 0; dd < 16; ++dd) { const float t = s[o * 16 + dd]; sq += t * t; }
    const float f = (sq / (1.f + sq)) * rsqrtf(sq + EPS_SQ);
#pragma unroll
    for (int dd = 0; dd < 16; ++dd)
      vout[((size_t)b * NOUT + o) * 16 + dd] = f * s[o * 16 + dd];
  }
}

// ---------------------------------------------------------------------------
// prev_t[b,p,e] = out0[b,p,:] @ dim_transform1[:,e]
// ---------------------------------------------------------------------------
__global__ void k_prev_t(const float* __restrict__ out0, const float* __restrict__ dt,
                         float* __restrict__ prev_t) {
  const int idx = blockIdx.x * blockDim.x + threadIdx.x;
  if (idx >= 128 * 5 * 16) return;
  const int e = idx % 16, t = idx / 16;
  float a = 0.f;
#pragma unroll
  for (int d = 0; d < 16; ++d) a += out0[(size_t)t * 16 + d] * dt[d * 16 + e];
  prev_t[idx] = a;
}

// Hierarchical agreement gate: votes1 *= sigmoid(sum_p agree_p * h_gate[o,p])
__global__ __launch_bounds__(256) void k_gate1(float* __restrict__ votes1,
                                               const float* __restrict__ prev_t,
                                               const float* __restrict__ h_gate) {
  const size_t idx = (size_t)blockIdx.x * blockDim.x + threadIdx.x;
  if (idx >= (size_t)128 * 581 * 25) return;
  const int o = (int)(idx % 25);
  const size_t t = idx / 25;
  const int b = (int)(t / 581);
  float* vr = &votes1[idx * 16];
  float row[16];
#pragma unroll
  for (int d = 0; d < 16; ++d) row[d] = vr[d];
  float gsum = 0.f;
#pragma unroll
  for (int p = 0; p < 5; ++p) {
    const float* pt = &prev_t[((size_t)b * 5 + p) * 16];
    float a = 0.f;
#pragma unroll
    for (int d = 0; d < 16; ++d) a += row[d] * pt[d];
    gsum += a * h_gate[o * 5 + p];
  }
  const float cons = 1.f / (1.f + __expf(-gsum));
#pragma unroll
  for (int d = 0; d < 16; ++d) vr[d] = row[d] * cons;
}

// soft[b,p,o] = 0.8*sigmoid(10*(tax0[p,o]*softmax_p(|out0|) - 0.5)) + 0.1
__global__ void k_soft(const float* __restrict__ out0, const float* __restrict__ tax0,
                       float* __restrict__ soft) {
  const int b = blockIdx.x;
  __shared__ float act[5];
  if (threadIdx.x < 5) {
    const float* op = &out0[((size_t)b * 5 + threadIdx.x) * 16];
    float sq = 0.f;
#pragma unroll
    for (int d = 0; d < 16; ++d) sq += op[d] * op[d];
    act[threadIdx.x] = sqrtf(sq + EPS_SQ);
  }
  __syncthreads();
  if (threadIdx.x == 0) {
    float mx = act[0];
    for (int p = 1; p < 5; ++p) mx = fmaxf(mx, act[p]);
    float e[5], sum = 0.f;
    for (int p = 0; p < 5; ++p) { e[p] = __expf(act[p] - mx); sum += e[p]; }
    for (int p = 0; p < 5; ++p) act[p] = e[p] / sum;
  }
  __syncthreads();
  if (threadIdx.x < 125) {
    const int p = threadIdx.x / 25;
    const float wt = tax0[threadIdx.x] * act[p];
    soft[(size_t)b * 125 + threadIdx.x] =
        0.8f / (1.f + __expf(-10.f * (wt - 0.5f))) + 0.1f;
  }
}

// tmp = v + bo  (residual + output bias; initializes the MHA accumulator)
__global__ void k_bias_res(const float* __restrict__ v, const float* __restrict__ bo,
                           float* __restrict__ tmp, int total) {
  const int idx = blockIdx.x * blockDim.x + threadIdx.x;
  if (idx >= total) return;
  tmp[idx] = v[idx] + bo[idx & 15];
}

// ---------------------------------------------------------------------------
// Multi-head self-attention, one block per (batch, head); everything in LDS.
// wq/wk/wv: [16, 16h, 32k]; wo: [16h, 32k, 16]. Accumulate via global atomics.
// ---------------------------------------------------------------------------
template <int N>
__global__ __launch_bounds__(256) void k_mha_head(const float* __restrict__ xin,
                                                  const float* __restrict__ wq,
                                                  const float* __restrict__ bq,
                                                  const float* __restrict__ wk,
                                                  const float* __restrict__ bk,
                                                  const float* __restrict__ wv,
                                                  const float* __restrict__ bv,
                                                  const float* __restrict__ wo,
                                                  float* __restrict__ acc) {
  const int b = blockIdx.x >> 4;
  const int h = blockIdx.x & 15;
  __shared__ float xs[N * 16];
  __shared__ float qs[N * 32], ks[N * 32], vs[N * 32], os[N * 32];
  __shared__ float att[N * N];

  for (int t = threadIdx.x; t < N * 16; t += blockDim.x)
    xs[t] = xin[(size_t)b * N * 16 + t];
  __syncthreads();

  for (int t = threadIdx.x; t < N * 32; t += blockDim.x) {
    const int r = t / 32, c = t % 32;
    float aq = bq[h * 32 + c], ak = bk[h * 32 + c], av = bv[h * 32 + c];
#pragma unroll
    for (int d = 0; d < 16; ++d) {
      const float xv = xs[r * 16 + d];
      const size_t wi = ((size_t)d * 16 + h) * 32 + c;
      aq += xv * wq[wi];
      ak += xv * wk[wi];
      av += xv * wv[wi];
    }
    qs[t] = aq; ks[t] = ak; vs[t] = av;
  }
  __syncthreads();

  for (int t = threadIdx.x; t < N * N; t += blockDim.x) {
    const int r = t / N, m = t % N;
    float a = 0.f;
#pragma unroll
    for (int c = 0; c < 32; ++c) a += qs[r * 32 + c] * ks[m * 32 + c];
    att[t] = a * 0.17677669529663687f;    // 1/sqrt(32)
  }
  __syncthreads();

  if (threadIdx.x < N) {                  // row softmax
    float* ar = &att[threadIdx.x * N];
    float mx = ar[0];
    for (int m = 1; m < N; ++m) mx = fmaxf(mx, ar[m]);
    float sum = 0.f;
    for (int m = 0; m < N; ++m) { ar[m] = __expf(ar[m] - mx); sum += ar[m]; }
    const float inv = 1.f / sum;
    for (int m = 0; m < N; ++m) ar[m] *= inv;
  }
  __syncthreads();

  for (int t = threadIdx.x; t < N * 32; t += blockDim.x) {
    const int r = t / 32, c = t % 32;
    float a = 0.f;
    for (int m = 0; m < N; ++m) a += att[r * N + m] * vs[m * 32 + c];
    os[t] = a;
  }
  __syncthreads();

  for (int t = threadIdx.x; t < N * 16; t += blockDim.x) {
    const int r = t / 16, dd = t % 16;
    float a = 0.f;
#pragma unroll
    for (int c = 0; c < 32; ++c) a += os[r * 32 + c] * wo[((size_t)h * 32 + c) * 16 + dd];
    atomicAdd(&acc[(size_t)b * N * 16 + t], a);
  }
}

// LayerNorm over the 16-dim capsule axis; one thread per row.
__global__ void k_layernorm(const float* __restrict__ xin, const float* __restrict__ g,
                            const float* __restrict__ bt, float* __restrict__ out,
                            int rows) {
  const int r = blockIdx.x * blockDim.x + threadIdx.x;
  if (r >= rows) return;
  const float* xr = &xin[(size_t)r * 16];
  float m = 0.f;
#pragma unroll
  for (int d = 0; d < 16; ++d) m += xr[d];
  m *= (1.f / 16.f);
  float var = 0.f;
#pragma unroll
  for (int d = 0; d < 16; ++d) { const float t = xr[d] - m; var += t * t; }
  var *= (1.f / 16.f);
  const float inv = rsqrtf(var + LN_EPS);
#pragma unroll
  for (int d = 0; d < 16; ++d)
    out[(size_t)r * 16 + d] = (xr[d] - m) * inv * g[d] + bt[d];
}

// ---------------------------------------------------------------------------
// Host-side orchestration
// ---------------------------------------------------------------------------
extern "C" void kernel_launch(void* const* d_in, const int* in_sizes, int n_in,
                              void* d_out, int out_size, void* d_ws, size_t ws_size,
                              hipStream_t stream) {
  // Inputs in setup_inputs() insertion order (params dict flattened in order):
  const float* x    = (const float*)d_in[0];   // [128,1152,8]
  const float* W0   = (const float*)d_in[1];   // [1152,5,16,8]
  const float* W1   = (const float*)d_in[2];   // [581,25,16,16]
  const float* hg1  = (const float*)d_in[3];   // [25,5]
  const float* dimt = (const float*)d_in[4];   // [16,16]
  // mha0: wq bq wk bk wv bv wo bo  -> d_in[5..12]
  // mha1: wq bq wk bk wv bv wo bo  -> d_in[13..20]
  const float* ln0g = (const float*)d_in[21];
  const float* ln0b = (const float*)d_in[22];
  const float* ln1g = (const float*)d_in[23];
  const float* ln1b = (const float*)d_in[24];
  const float* tax0 = (const float*)d_in[25];

  float* out0 = (float*)d_out;                 // [128,5,16]
  float* out1 = (float*)d_out + 128 * 5 * 16;  // [128,25,16]

  // Workspace layout (floats)
  float* ws = (float*)d_ws;
  float* votes0 = ws;                                   // 11,796,480
  float* votes1 = votes0 + (size_t)128 * 1152 * 5 * 16; // 29,747,200
  float* raw0   = votes1 + (size_t)128 * 581 * 25 * 16; //    737,280
  float* raw1   = raw0   + (size_t)128 * 1152 * 5;      //  1,859,200
  float* v0buf  = raw1   + (size_t)128 * 581 * 25;      //     10,240
  float* v1buf  = v0buf  + 128 * 5 * 16;                //     51,200
  float* prevt  = v1buf  + 128 * 25 * 16;               //     10,240
  float* soft   = prevt  + 128 * 5 * 16;                //     16,000
  float* tmp0   = soft   + 128 * 5 * 25;                //     10,240
  float* tmp1   = tmp0   + 128 * 5 * 16;                //     51,200

  // ---------------- Level 0 ----------------
  k_votes0<<<1152, 256, 0, stream>>>(x, W0, votes0);
  k_route_s0<1152, 5><<<80, 256, 0, stream>>>(votes0, v0buf);
  k_raw<1152, 5><<<(128 * 1152 * 5 + 255) / 256, 256, 0, stream>>>(votes0, v0buf, raw0);
  k_route_final<1152, 5, false><<<128, 256, 0, stream>>>(votes0, raw0, nullptr, v0buf);
  k_bias_res<<<(10240 + 255) / 256, 256, 0, stream>>>(v0buf, (const float*)d_in[12], tmp0, 10240);
  k_mha_head<5><<<128 * 16, 256, 0, stream>>>(v0buf,
      (const float*)d_in[5], (const float*)d_in[6], (const float*)d_in[7],
      (const float*)d_in[8], (const float*)d_in[9], (const float*)d_in[10],
      (const float*)d_in[11], tmp0);
  k_layernorm<<<(640 + 255) / 256, 256, 0, stream>>>(tmp0, ln0g, ln0b, out0, 640);

  // ---------------- Level 1 ----------------
  k_prev_t<<<(10240 + 255) / 256, 256, 0, stream>>>(out0, dimt, prevt);
  k_soft<<<128, 128, 0, stream>>>(out0, tax0, soft);
  k_votes1<<<581, 256, 0, stream>>>(x, out0, W1, votes1);
  k_gate1<<<(int)(((size_t)128 * 581 * 25 + 255) / 256), 256, 0, stream>>>(votes1, prevt, hg1);
  k_route_s0<581, 25><<<400, 256, 0, stream>>>(votes1, v1buf);
  k_raw<581, 25><<<(int)(((size_t)128 * 581 * 25 + 255) / 256), 256, 0, stream>>>(votes1, v1buf, raw1);
  k_route_final<581, 25, true><<<128, 256, 0, stream>>>(votes1, raw1, soft, v1buf);
  k_bias_res<<<(51200 + 255) / 256, 256, 0, stream>>>(v1buf, (const float*)d_in[20], tmp1, 51200);
  k_mha_head<25><<<128 * 16, 256, 0, stream>>>(v1buf,
      (const float*)d_in[13], (const float*)d_in[14], (const float*)d_in[15],
      (const float*)d_in[16], (const float*)d_in[17], (const float*)d_in[18],
      (const float*)d_in[19], tmp1);
  k_layernorm<<<(3200 + 255) / 256, 256, 0, stream>>>(tmp1, ln1g, ln1b, out1, 3200);

  (void)in_sizes; (void)n_in; (void)out_size; (void)ws_size;
}
